// FlashAttention_52201032515788
// MI455X (gfx1250) — matrix operand order repaired and checked by reading, not compiled, not run on hardware
//
#include <hip/hip_runtime.h>
#include <hip/hip_bf16.h>
#include <stdint.h>

// ---------------------------------------------------------------------------
// B=4, S=4096, D=2048, H=16, HEAD_DIM=384, D3=128
// Pipeline:
//   prep : x -> bf16; Wqkv,Wout -> bf16 transposed [n][k]
//   k1   : qkv[16384][6144](bf16) = xbf @ WqkvT + bqkv ; also scatter V into
//          vT[b][h][d][s] (pre-transposed for flash)
//   k2   : causal flash attention (TDM-staged Q/K/V tiles)
//   k3   : y[16384][2048](f32) = attn(bf16) @ WoutT + bout
// All matrix math: v_wmma_f32_16x16x32_bf16 with f32 accumulation.
// ---------------------------------------------------------------------------

typedef __attribute__((ext_vector_type(16))) __bf16 v16bf;
typedef __attribute__((ext_vector_type(8)))  float  v8f;
typedef unsigned int u32x4 __attribute__((ext_vector_type(4)));
typedef int          i32x8 __attribute__((ext_vector_type(8)));
typedef int          i32x4 __attribute__((ext_vector_type(4)));

#if defined(__AMDGCN__) && (__clang_major__ < 23) && \
    __has_builtin(__builtin_amdgcn_tensor_load_to_lds)
#define USE_TDM 1
#else
#define USE_TDM 0
#endif

union AFrag {
    v16bf v;
    uint4 q[2];
};

__device__ __forceinline__ unsigned short f2bf(float f) {
    unsigned int u = __builtin_bit_cast(unsigned int, f);
    u += 0x7fffu + ((u >> 16) & 1u);          // round-to-nearest-even
    return (unsigned short)(u >> 16);
}

__device__ __forceinline__ unsigned int pack2bf(float lo, float hi) {
#if defined(__AMDGCN__) && __has_builtin(__builtin_amdgcn_cvt_pk_bf16_f32)
    auto r = __builtin_amdgcn_cvt_pk_bf16_f32(lo, hi);
    return __builtin_bit_cast(unsigned int, r);
#else
    return (unsigned)f2bf(lo) | ((unsigned)f2bf(hi) << 16);
#endif
}

__device__ __forceinline__ v8f vzero8() {
    v8f z;
#pragma unroll
    for (int e = 0; e < 8; ++e) z[e] = 0.f;
    return z;
}

// A-matrix 16x32 bf16 fragment (ISA 7.12.2); tile row-major in LDS.
__device__ __forceinline__ void load_a_frag(const unsigned short* base, int stride,
                                            int k0, AFrag& f) {
    const int lane = threadIdx.x & 31;
    const int row  = lane & 15;
    const int ko   = (lane >> 4) * 8;
    const unsigned short* p = base + row * stride + k0 + ko;
    f.q[0] = *(const uint4*)(p);
    f.q[1] = *(const uint4*)(p + 16);
}

// B-matrix 32x16 bf16 fragment; tile stored K-contiguous per column in LDS.
__device__ __forceinline__ void load_b_frag(const unsigned short* base, int stride,
                                            int k0, AFrag& f) {
    const int lane = threadIdx.x & 31;
    const int col  = lane & 15;
    const int kb   = (lane >> 4) * 8;
    const unsigned short* p = base + col * stride + k0 + kb;
    f.q[0] = *(const uint4*)(p);
    f.q[1] = *(const uint4*)(p + 16);
}

__device__ __forceinline__ v8f wmma_bf16(const AFrag& a, const AFrag& b, v8f c) {
    return __builtin_amdgcn_wmma_f32_16x16x32_bf16(
        false, a.v, false, b.v, (short)0, c, false, false);
}

#if USE_TDM
// Issue one 2D-tile TDM load (D# per ISA 8.3/8.4): tile_h rows x tile_w elems
// (2-byte), row stride `stride` elems, packed contiguously at lds_addr.
__device__ __forceinline__ void tdm_load_2d(unsigned lds_addr, const void* gptr,
                                            unsigned tile_w, unsigned tile_h,
                                            unsigned stride) {
    const size_t ga = (size_t)gptr;
    u32x4 g0;
    g0[0] = 1u;                                          // count=1 user D#
    g0[1] = lds_addr;
    g0[2] = (unsigned)(ga & 0xffffffffu);
    g0[3] = (unsigned)((ga >> 32) & 0x01ffffffu) | (2u << 30);  // type=2
    i32x8 g1;
    g1[0] = (int)(1u << 16);                             // data_size = 2 bytes
    g1[1] = (int)((tile_w & 0xffffu) << 16);             // tensor_dim0
    g1[2] = (int)((tile_w >> 16) | ((tile_h & 0xffffu) << 16)); // tensor_dim1
    g1[3] = (int)((tile_h >> 16) | (tile_w << 16));      // tile_dim0
    g1[4] = (int)(tile_h);                               // tile_dim1 (dim2=0)
    g1[5] = (int)(stride);                               // tensor_dim0_stride
    g1[6] = 0;
    g1[7] = 0;
    i32x4 z4 = (i32x4)0;
    __builtin_amdgcn_tensor_load_to_lds(g0, g1, z4, z4, 0);
}
#endif

// ---------------------------------------------------------------------------
// Prep kernels: one-shot conversions so GEMM hot loops are copy-only.
// ---------------------------------------------------------------------------
__global__ __launch_bounds__(256) void cvt_bf16_kernel(
    const float4* __restrict__ in, uint2* __restrict__ out) {
    int i = blockIdx.x * 256 + threadIdx.x;
    float4 f = in[i];
    uint2 p;
    p.x = pack2bf(f.x, f.y);
    p.y = pack2bf(f.z, f.w);
    out[i] = p;
}

// in f32 [R][C] -> out bf16 [C][R]; grid (C/32, R/32), block (32,8)
__global__ __launch_bounds__(256) void transpose_cvt_kernel(
    const float* __restrict__ in, unsigned short* __restrict__ out,
    int R, int C) {
    __shared__ float tile[32][33];
    const int n0 = blockIdx.x * 32, k0 = blockIdx.y * 32;
    const int tx = threadIdx.x, ty = threadIdx.y;
#pragma unroll
    for (int i = 0; i < 4; ++i)
        tile[ty + i * 8][tx] = in[(size_t)(k0 + ty + i * 8) * C + n0 + tx];
    __syncthreads();
#pragma unroll
    for (int i = 0; i < 4; ++i)
        out[(size_t)(n0 + ty + i * 8) * R + k0 + tx] = f2bf(tile[tx][ty + i * 8]);
}

// ---------------------------------------------------------------------------
// Shared GEMM tile staging: both operands are bf16, K-contiguous, stride 2048.
// ---------------------------------------------------------------------------
__device__ __forceinline__ void tile_issue(const unsigned short* __restrict__ src,
                                           int r0, int k0, int tid, uint4 (&q)[2]) {
#pragma unroll
    for (int it = 0; it < 2; ++it) {
        int lin = tid + it * 256;
        q[it] = *(const uint4*)(src + (size_t)(r0 + (lin >> 2)) * 2048 + k0 + (lin & 3) * 8);
    }
}

__device__ __forceinline__ void tile_commit(unsigned short* dst, int tid,
                                            const uint4 (&q)[2]) {
#pragma unroll
    for (int it = 0; it < 2; ++it) {
        int lin = tid + it * 256;
        *(uint4*)&dst[(lin >> 2) * 32 + (lin & 3) * 8] = q[it];
    }
}

// ---------------------------------------------------------------------------
// Kernel 1: qkv = xbf @ WqkvT + bqkv ; V part also scattered into vT[b][h][d][s]
// ---------------------------------------------------------------------------
__global__ __launch_bounds__(256)
__attribute__((amdgpu_waves_per_eu(1)))      // no spills: grid sets occupancy
void qkv_gemm_kernel(
    const unsigned short* __restrict__ xbf, const unsigned short* __restrict__ wT,
    const float* __restrict__ bias, unsigned short* __restrict__ qkv,
    unsigned short* __restrict__ vT)
{
    constexpr int K = 2048, N = 6144;
    __shared__ __align__(16) unsigned short As[2][128 * 32];
    __shared__ __align__(16) unsigned short Bt[2][128 * 32];
    const int tid  = threadIdx.x;
    const int lane = tid & 31;
    const int wave = tid >> 5;
    const int n0 = blockIdx.x * 128;
    const int m0 = blockIdx.y * 128;
    const int wm = (wave & 3) * 32;
    const int wn = (wave >> 2) * 64;

    v8f acc[2][4];
#pragma unroll
    for (int i = 0; i < 2; ++i)
#pragma unroll
        for (int j = 0; j < 4; ++j) acc[i][j] = vzero8();

    uint4 qa[2], qb[2];
    tile_issue(xbf, m0, 0, tid, qa);
    tile_issue(wT,  n0, 0, tid, qb);
    tile_commit(&As[0][0], tid, qa);
    tile_commit(&Bt[0][0], tid, qb);
    __syncthreads();

    int p = 0;
    for (int k0 = 0; k0 < K; k0 += 32) {
        const bool has_next = (k0 + 32 < K);
        if (has_next) {                      // HBM loads in flight over the WMMAs
            tile_issue(xbf, m0, k0 + 32, tid, qa);
            tile_issue(wT,  n0, k0 + 32, tid, qb);
        }
        if (k0 + 64 < K)
            __builtin_prefetch(xbf + (size_t)(m0 + (tid >> 2)) * K + (k0 + 64) + (tid & 3) * 8, 0, 3);

        AFrag a0, a1, b;
        load_a_frag(&As[p][(wm + 0) * 32], 32, 0, a0);
        load_a_frag(&As[p][(wm + 16) * 32], 32, 0, a1);
#pragma unroll
        for (int nt = 0; nt < 4; ++nt) {
            load_b_frag(&Bt[p][(wn + nt * 16) * 32], 32, 0, b);
            acc[0][nt] = wmma_bf16(a0, b, acc[0][nt]);
            acc[1][nt] = wmma_bf16(a1, b, acc[1][nt]);
        }
        if (has_next) {
            tile_commit(&As[p ^ 1][0], tid, qa);
            tile_commit(&Bt[p ^ 1][0], tid, qb);
        }
        __syncthreads();
        p ^= 1;
    }

    const int colb = lane & 15;
    const int hi   = lane >> 4;
#pragma unroll
    for (int nt = 0; nt < 4; ++nt) {
        const int col = n0 + wn + nt * 16 + colb;
        const float bv = bias[col];
        const int h  = col / 384;
        const int dd = col - h * 384;
#pragma unroll
        for (int mt = 0; mt < 2; ++mt)
#pragma unroll
            for (int r = 0; r < 8; ++r) {
                const int row = m0 + wm + mt * 16 + r + 8 * hi;
                const unsigned short bf = f2bf(acc[mt][nt][r] + bv);
                qkv[(size_t)row * N + col] = bf;
                if (dd >= 256) {                       // V -> vT[b][h][d][s]
                    const int bb = row >> 12, s = row & 4095;
                    vT[(((size_t)bb * 16 + h) * 128 + (dd - 256)) * 4096 + s] = bf;
                }
            }
    }
}

// ---------------------------------------------------------------------------
// Kernel 2: causal flash attention. 64-row Q block / 128 threads (4 waves,
// one 16-row strip each). Q/K from qkv[B,S,H,384]; V from vT[b][h][d][s].
// All tiles TDM-staged when available.
// ---------------------------------------------------------------------------
__global__ __launch_bounds__(128)
__attribute__((amdgpu_waves_per_eu(1)))
void flash_attn_kernel(
    const unsigned short* __restrict__ qkv, const unsigned short* __restrict__ vTg,
    unsigned short* __restrict__ attn)
{
    constexpr int S = 4096, H = 16, QKV = 6144, DH = 128;
    __shared__ __align__(16) unsigned short Qs[64 * 128];
    __shared__ __align__(16) unsigned short Ksh[64 * 128];
    __shared__ __align__(16) unsigned short Vt[128 * 64];    // [d][seq]
    __shared__ __align__(16) unsigned short Ps[4][16 * 64];

    const int tid  = threadIdx.x;
    const int lane = tid & 31;
    const int wave = tid >> 5;
    const int i0 = blockIdx.x * 64;
    const int b  = blockIdx.y >> 4;
    const int h  = blockIdx.y & 15;
    const size_t base  = (size_t)b * S * QKV + (size_t)h * 384;
    const size_t vbase = (((size_t)b * H + h) * 128) * 4096;

#if USE_TDM
    const int wv = __builtin_amdgcn_readfirstlane((int)(threadIdx.x >> 5));
    tdm_load_2d((unsigned)(size_t)(void*)&Qs[(wv * 16) * 128],
                qkv + base + (size_t)(i0 + wv * 16) * QKV, 128, 16, QKV);
    __builtin_amdgcn_s_wait_tensorcnt(0);
#else
#pragma unroll
    for (int it = 0; it < 8; ++it) {        // stage Q (64x128 bf16)
        int lin = tid + it * 128;
        int row = lin >> 4;
        int c8  = (lin & 15) * 8;
        *(uint4*)&Qs[row * 128 + c8] =
            *(const uint4*)(qkv + base + (size_t)(i0 + row) * QKV + c8);
    }
#endif
    __syncthreads();

    AFrag qf[4];
#pragma unroll
    for (int k = 0; k < 4; ++k)
        load_a_frag(&Qs[(wave * 16) * 128], 128, k * 32, qf[k]);

    float m_r[8], l_r[8];
    v8f O[8];
#pragma unroll
    for (int r = 0; r < 8; ++r) { m_r[r] = -3.0e38f; l_r[r] = 0.f; }
#pragma unroll
    for (int dt = 0; dt < 8; ++dt) O[dt] = vzero8();

    const float scale = 0.08838834764831845f;   // 1/sqrt(128)
    const int nj = (i0 >> 6) + 1;

    for (int jb = 0; jb < nj; ++jb) {
        const int j0 = jb * 64;
        __syncthreads();   // previous Ksh/Vt fully consumed

#if USE_TDM
        // Each wave DMA-loads a 16x128 K slice and a 32x64 V slice.
        tdm_load_2d((unsigned)(size_t)(void*)&Ksh[(wv * 16) * 128],
                    qkv + base + (size_t)(j0 + wv * 16) * QKV + 128, 128, 16, QKV);
        tdm_load_2d((unsigned)(size_t)(void*)&Vt[(wv * 32) * 64],
                    vTg + vbase + (size_t)(wv * 32) * 4096 + j0, 64, 32, 4096);
        __builtin_amdgcn_s_wait_tensorcnt(0);
#else
#pragma unroll
        for (int it = 0; it < 8; ++it) {    // K stage
            int lin = tid + it * 128;
            int row = lin >> 4;
            int c8  = (lin & 15) * 8;
            *(uint4*)&Ksh[row * 128 + c8] =
                *(const uint4*)(qkv + base + (size_t)(j0 + row) * QKV + 128 + c8);
        }
#pragma unroll
        for (int it = 0; it < 8; ++it) {    // V stage (already [d][s] in memory)
            int lin = tid + it * 128;
            int d  = lin >> 3;
            int c8 = (lin & 7) * 8;
            *(uint4*)&Vt[d * 64 + c8] =
                *(const uint4*)(vTg + vbase + (size_t)d * 4096 + j0 + c8);
        }
#endif
        __syncthreads();

        // S strip (16x64) = Qstrip(16x128) . K^T
        v8f sacc[4];
#pragma unroll
        for (int nt = 0; nt < 4; ++nt) {
            sacc[nt] = vzero8();
#pragma unroll
            for (int k = 0; k < 4; ++k) {
                AFrag bf;
                load_b_frag(&Ksh[(nt * 16) * 128], 128, k * 32, bf);
                sacc[nt] = wmma_bf16(qf[k], bf, sacc[nt]);
            }
        }

        // online softmax (acc layout: row = VGPR idx + 8*(lane>=16), col = lane&15)
        float pv[4][8];
#pragma unroll
        for (int r = 0; r < 8; ++r) {
            const int rowg = i0 + wave * 16 + r + 8 * (lane >> 4);
            float mx = -3.0e38f;
#pragma unroll
            for (int nt = 0; nt < 4; ++nt) {
                const int colg = j0 + nt * 16 + (lane & 15);
                float vsc = sacc[nt][r] * scale;
                if (colg > rowg) vsc = -3.0e38f;   // causal mask
                pv[nt][r] = vsc;
                mx = fmaxf(mx, vsc);
            }
#pragma unroll
            for (int msk = 1; msk <= 8; msk <<= 1)
                mx = fmaxf(mx, __shfl_xor(mx, msk, 32));
            const float mnew = fmaxf(m_r[r], mx);
            const float corr = __expf(m_r[r] - mnew);
            m_r[r] = mnew;
            float rs = 0.f;
#pragma unroll
            for (int nt = 0; nt < 4; ++nt) {
                float pe = __expf(pv[nt][r] - mnew);
                pv[nt][r] = pe;
                rs += pe;
            }
#pragma unroll
            for (int msk = 1; msk <= 8; msk <<= 1)
                rs += __shfl_xor(rs, msk, 32);
            l_r[r] = l_r[r] * corr + rs;
#pragma unroll
            for (int dt = 0; dt < 8; ++dt) O[dt][r] *= corr;
        }

        // P: C layout -> LDS -> A layout (same-wave DS ordering, no barrier)
        unsigned short* P = &Ps[wave][0];
#pragma unroll
        for (int nt = 0; nt < 4; ++nt)
#pragma unroll
            for (int r = 0; r < 8; ++r)
                P[(r + 8 * (lane >> 4)) * 64 + nt * 16 + (lane & 15)] = f2bf(pv[nt][r]);

        AFrag pf0, pf1;
        load_a_frag(P, 64, 0,  pf0);
        load_a_frag(P, 64, 32, pf1);

        // O(16x128) += P(16x64) . V(64x128)
#pragma unroll
        for (int dt = 0; dt < 8; ++dt) {
            AFrag bf;
            load_b_frag(&Vt[(dt * 16) * 64], 64, 0, bf);
            O[dt] = wmma_bf16(pf0, bf, O[dt]);
            load_b_frag(&Vt[(dt * 16) * 64], 64, 32, bf);
            O[dt] = wmma_bf16(pf1, bf, O[dt]);
        }
    }

#pragma unroll
    for (int dt = 0; dt < 8; ++dt)
#pragma unroll
        for (int r = 0; r < 8; ++r) {
            int sg = i0 + wave * 16 + r + 8 * (lane >> 4);
            float o = O[dt][r] / l_r[r];
            attn[(((size_t)b * S + sg) * H + h) * DH + dt * 16 + (lane & 15)] = f2bf(o);
        }
}

// ---------------------------------------------------------------------------
// Kernel 3: y (f32) = attn(bf16) @ WoutT(bf16 [n][k]) + bout
// ---------------------------------------------------------------------------
__global__ __launch_bounds__(256)
__attribute__((amdgpu_waves_per_eu(1)))
void out_gemm_kernel(
    const unsigned short* __restrict__ a, const unsigned short* __restrict__ wT,
    const float* __restrict__ bias, float* __restrict__ out)
{
    constexpr int K = 2048, N = 2048;
    __shared__ __align__(16) unsigned short As[2][128 * 32];
    __shared__ __align__(16) unsigned short Bt[2][128 * 32];
    const int tid  = threadIdx.x;
    const int lane = tid & 31;
    const int wave = tid >> 5;
    const int n0 = blockIdx.x * 128;
    const int m0 = blockIdx.y * 128;
    const int wm = (wave & 3) * 32;
    const int wn = (wave >> 2) * 64;

    v8f acc[2][4];
#pragma unroll
    for (int i = 0; i < 2; ++i)
#pragma unroll
        for (int j = 0; j < 4; ++j) acc[i][j] = vzero8();

    uint4 qa[2], qb[2];
    tile_issue(a,  m0, 0, tid, qa);
    tile_issue(wT, n0, 0, tid, qb);
    tile_commit(&As[0][0], tid, qa);
    tile_commit(&Bt[0][0], tid, qb);
    __syncthreads();

    int p = 0;
    for (int k0 = 0; k0 < K; k0 += 32) {
        const bool has_next = (k0 + 32 < K);
        if (has_next) {
            tile_issue(a,  m0, k0 + 32, tid, qa);
            tile_issue(wT, n0, k0 + 32, tid, qb);
        }
        if (k0 + 64 < K)
            __builtin_prefetch(wT + (size_t)(n0 + (tid >> 2)) * K + (k0 + 64) + (tid & 3) * 8, 0, 3);

        AFrag a0, a1, b;
        load_a_frag(&As[p][(wm + 0) * 32], 32, 0, a0);
        load_a_frag(&As[p][(wm + 16) * 32], 32, 0, a1);
#pragma unroll
        for (int nt = 0; nt < 4; ++nt) {
            load_b_frag(&Bt[p][(wn + nt * 16) * 32], 32, 0, b);
            acc[0][nt] = wmma_bf16(a0, b, acc[0][nt]);
            acc[1][nt] = wmma_bf16(a1, b, acc[1][nt]);
        }
        if (has_next) {
            tile_commit(&As[p ^ 1][0], tid, qa);
            tile_commit(&Bt[p ^ 1][0], tid, qb);
        }
        __syncthreads();
        p ^= 1;
    }

    const int colb = lane & 15;
    const int hi   = lane >> 4;
#pragma unroll
    for (int nt = 0; nt < 4; ++nt) {
        int col = n0 + wn + nt * 16 + colb;
        float bv = bias[col];
#pragma unroll
        for (int mt = 0; mt < 2; ++mt)
#pragma unroll
            for (int r = 0; r < 8; ++r) {
                int row = m0 + wm + mt * 16 + r + 8 * hi;
                out[(size_t)row * N + col] = acc[mt][nt][r] + bv;
            }
    }
}

// ---------------------------------------------------------------------------
// Workspace (bytes):
//   qkv   @ 0          : 16384*6144*2 = 201326592
//   vT    @ 201326592  : 4*16*128*4096*2 = 67108864
//   xbf   @ 268435456  : 16384*2048*2 = 67108864   (attn aliases this later)
//   WqkvT @ 335544320  : 6144*2048*2 = 25165824
//   WoutT @ 360710144  : 2048*2048*2 = 8388608
// ---------------------------------------------------------------------------
extern "C" void kernel_launch(void* const* d_in, const int* in_sizes, int n_in,
                              void* d_out, int out_size, void* d_ws, size_t ws_size,
                              hipStream_t stream) {
    const float* x    = (const float*)d_in[0];
    const float* Wqkv = (const float*)d_in[1];
    const float* bqkv = (const float*)d_in[2];
    const float* Wout = (const float*)d_in[3];
    const float* bout = (const float*)d_in[4];
    float* out = (float*)d_out;

    char* ws = (char*)d_ws;
    unsigned short* qkv   = (unsigned short*)(ws);
    unsigned short* vT    = (unsigned short*)(ws + 201326592ull);
    unsigned short* xbf   = (unsigned short*)(ws + 268435456ull);
    unsigned short* attn  = xbf;   // aliases xbf (dead after kernel 1)
    unsigned short* WqkvT = (unsigned short*)(ws + 335544320ull);
    unsigned short* WoutT = (unsigned short*)(ws + 360710144ull);

    // prep: one-shot conversions (bandwidth-trivial vs 23.3 TB/s)
    cvt_bf16_kernel<<<32768, 256, 0, stream>>>((const float4*)x, (uint2*)xbf);
    transpose_cvt_kernel<<<dim3(192, 64), dim3(32, 8), 0, stream>>>(Wqkv, WqkvT, 2048, 6144);
    transpose_cvt_kernel<<<dim3(64, 64), dim3(32, 8), 0, stream>>>(Wout, WoutT, 2048, 2048);

    qkv_gemm_kernel<<<dim3(48, 128), 256, 0, stream>>>(xbf, WqkvT, bqkv, qkv, vT);
    flash_attn_kernel<<<dim3(64, 64), 128, 0, stream>>>(qkv, vT, attn);
    out_gemm_kernel<<<dim3(16, 128), 256, 0, stream>>>(attn, WoutT, bout, out);
}
